// FeatureGenNet_22067541967677
// MI455X (gfx1250) — matrix operand order repaired and checked by reading, not compiled, not run on hardware
//
#include <hip/hip_runtime.h>

// ---------------------------------------------------------------------------
// CDNA5 (gfx1250) implementation of FeatureGenNet forward pass.
// Convs = implicit-GEMM on V_WMMA_F32_16X16X32_F16 (wave32), 64x128 tile,
// software-pipelined double-buffered LDS, A tile via async global->LDS b128,
// B tile register-staged so global latency hides under the WMMAs.
// Activations live in global memory as f16; accumulation is f32.
// ---------------------------------------------------------------------------

typedef __attribute__((ext_vector_type(16))) _Float16 v16h;
typedef __attribute__((ext_vector_type(8)))  _Float16 v8h;
typedef __attribute__((ext_vector_type(8)))  float    v8f;

#define A_ST 40   // A LDS row stride in halves (80B: 16B-aligned rows)
#define B_ST 40   // B LDS row stride in halves

// LDS-offset of a shared-memory pointer (AS3 ptrtoint -> wave-relative offset)
#define FG_LDS_OFF(p) ((unsigned)(unsigned long long)(__attribute__((address_space(3))) const void*)(p))

// ---------------------------------------------------------------------------
__global__ void fg_cvt_f32_f16(const float* __restrict__ in, _Float16* __restrict__ out, int n) {
  int i = blockIdx.x * blockDim.x + threadIdx.x;
  if (i < n) out[i] = (_Float16)in[i];
}

// ---------------------------------------------------------------------------
__global__ void fg_bn_prep(const float* __restrict__ g, const float* __restrict__ b,
                           const float* __restrict__ m, const float* __restrict__ v,
                           float* __restrict__ scale, float* __restrict__ bias, int C) {
  int i = blockIdx.x * blockDim.x + threadIdx.x;
  if (i >= C) return;
  float inv = g[i] * rsqrtf(v[i] + 1e-5f);
  scale[i] = inv;
  bias[i]  = b[i] - m[i] * inv;
}

// ---------------------------------------------------------------------------
// Implicit-GEMM conv, fused BN + optional residual + optional ReLU.
//   M = Cout, N = T*Ho*Wo, K = Cin*KH*KW
//   Block: 256 threads = 8 waves. Tile: 64(M) x 128(N) x 32(K).
//   Wave (wm 0..1, wn 0..3) owns a 32x32 slab -> 4 WMMAs per K-step.
// Pipeline per chunk c (1 barrier each):
//   ds_store B(c) from regs; s_wait_asynccnt 0 (A(c) landed); barrier;
//   issue async A(c+1) + global-load B(c+1) into regs; fragments(c); 4x WMMA.
// ---------------------------------------------------------------------------
__global__ __launch_bounds__(256) void fg_conv_wmma(
    const _Float16* __restrict__ in, const _Float16* __restrict__ w,
    const float* __restrict__ scale, const float* __restrict__ bias,
    const _Float16* __restrict__ residual, _Float16* __restrict__ out,
    int T, int Cin, int H, int W, int Cout, int Ho, int Wo,
    int KH, int KW, int stride, int pad, int relu)
{
  __shared__ __attribute__((aligned(16))) _Float16 As[2][64 * A_ST];    // [m][k]
  __shared__ __attribute__((aligned(16))) _Float16 Bs[2][128 * B_ST];   // [n][k]

  const int tid  = threadIdx.x;
  const int n0   = blockIdx.x * 128;
  const int m0   = blockIdx.y * 64;
  const int HWo  = Ho * Wo;
  const int Ntot = T * HWo;
  const int Ktot = Cin * KH * KW;
  const int KK   = KH * KW;
  const int HWin = H * W;

  const int lane = tid & 31;
  const int wv   = tid >> 5;        // wave 0..7
  const int wm   = wv >> 2;         // 0..1  (M sub-block of 32)
  const int wn   = wv & 3;          // 0..3  (N sub-block of 32)
  const int half = lane >> 4;       // 0/1
  const int l16  = lane & 15;

  v8f acc00 = {}, acc01 = {}, acc10 = {}, acc11 = {};

  // Per-thread decode of the 16 B-tile columns this thread fills (K-invariant).
  int nb[16]; unsigned pk[16];
#pragma unroll
  for (int i = 0; i < 16; ++i) {
    int n = n0 + i * 8 + (tid >> 5);
    if (n < Ntot) {
      int t  = n / HWo;
      int r  = n - t * HWo;
      int oy = r / Wo;
      int ox = r - oy * Wo;
      nb[i] = t * Cin * HWin;
      pk[i] = (unsigned)(oy | (ox << 16));
    } else {
      nb[i] = -1; pk[i] = 0;
    }
  }
  const int kk   = tid & 31;
  const int arow = tid >> 2;          // 0..63 : A row for async b128
  const int acol = (tid & 3) * 8;     // halves within the 32-K slab

  // ---- pipeline stage helpers ----
  auto issueA = [&](int c, int buf) {
    int kc = c << 5;
    if (kc + 32 <= Ktot) {
      unsigned lds = FG_LDS_OFF(&As[buf][arow * A_ST + acol]);
      unsigned long long ga =
          (unsigned long long)(const void*)(w + (size_t)(m0 + arow) * Ktot + kc + acol);
      asm volatile("global_load_async_to_lds_b128 %0, %1, off"
                   :: "v"(lds), "v"(ga) : "memory");
    } else {   // partial tail chunk (stem only): manual zero-filled fill
#pragma unroll
      for (int i = 0; i < 8; ++i) {
        int m_l = i * 8 + (tid >> 5);
        int k   = kc + kk;
        As[buf][m_l * A_ST + kk] = (k < Ktot) ? w[(size_t)(m0 + m_l) * Ktot + k]
                                              : (_Float16)0.0f;
      }
    }
  };
  auto loadB = [&](int c, _Float16 (&br)[16]) {
    int k = (c << 5) + kk;
    bool kok = (k < Ktot);
    int ci = 0, ky = 0, kx = 0;
    if (kok) {
      ci = k / KK;
      int rr = k - ci * KK;
      ky = rr / KW;
      kx = rr - ky * KW;
    }
    int cbase = ci * HWin;
#pragma unroll
    for (int i = 0; i < 16; ++i) {
      _Float16 val = (_Float16)0.0f;
      if (kok && nb[i] >= 0) {
        int oy = (int)(pk[i] & 0xffffu);
        int ox = (int)(pk[i] >> 16);
        int iy = oy * stride - pad + ky;
        int ix = ox * stride - pad + kx;
        if (iy >= 0 && iy < H && ix >= 0 && ix < W)
          val = in[(size_t)nb[i] + cbase + iy * W + ix];
      }
      br[i] = val;
    }
  };
  auto storeB = [&](int buf, _Float16 (&br)[16]) {
#pragma unroll
    for (int i = 0; i < 16; ++i)
      Bs[buf][(i * 8 + (tid >> 5)) * B_ST + kk] = br[i];
  };
  auto compute = [&](int buf) {
    // Fragments: per-lane 8-half runs -> two ds_load_b128 per fragment.
    // A: lanes 0-15 hold K[half*8,+8) and K[16+half*8,+8); rows = m (l16-based)
    // B: lanes hold K[half*8,+8) and K[16+half*8,+8);      rows = n
    const int r0 = (wm * 32      + l16) * A_ST;
    const int r1 = (wm * 32 + 16 + l16) * A_ST;
    const int c0 = (wn * 32      + l16) * B_ST;
    const int c1 = (wn * 32 + 16 + l16) * B_ST;
    v8h alo0 = *(const v8h*)&As[buf][r0 + half * 8];
    v8h ahi0 = *(const v8h*)&As[buf][r0 + 16 + half * 8];
    v8h alo1 = *(const v8h*)&As[buf][r1 + half * 8];
    v8h ahi1 = *(const v8h*)&As[buf][r1 + 16 + half * 8];
    v8h blo0 = *(const v8h*)&Bs[buf][c0 + half * 8];
    v8h bhi0 = *(const v8h*)&Bs[buf][c0 + 16 + half * 8];
    v8h blo1 = *(const v8h*)&Bs[buf][c1 + half * 8];
    v8h bhi1 = *(const v8h*)&Bs[buf][c1 + 16 + half * 8];
    v16h a0 = __builtin_shufflevector(alo0, ahi0, 0,1,2,3,4,5,6,7,8,9,10,11,12,13,14,15);
    v16h a1 = __builtin_shufflevector(alo1, ahi1, 0,1,2,3,4,5,6,7,8,9,10,11,12,13,14,15);
    v16h b0 = __builtin_shufflevector(blo0, bhi0, 0,1,2,3,4,5,6,7,8,9,10,11,12,13,14,15);
    v16h b1 = __builtin_shufflevector(blo1, bhi1, 0,1,2,3,4,5,6,7,8,9,10,11,12,13,14,15);
    acc00 = __builtin_amdgcn_wmma_f32_16x16x32_f16(false, a0, false, b0, (short)0, acc00, false, false);
    acc01 = __builtin_amdgcn_wmma_f32_16x16x32_f16(false, a0, false, b1, (short)0, acc01, false, false);
    acc10 = __builtin_amdgcn_wmma_f32_16x16x32_f16(false, a1, false, b0, (short)0, acc10, false, false);
    acc11 = __builtin_amdgcn_wmma_f32_16x16x32_f16(false, a1, false, b1, (short)0, acc11, false, false);
  };

  // ---- software-pipelined K loop (double-buffered, 1 barrier/chunk) ----
  _Float16 breg0[16], breg1[16];
  const int nch = (Ktot + 31) >> 5;
  issueA(0, 0);
  loadB(0, breg0);
  int c = 0;
  while (true) {
    // even step: chunk c in buffer 0
    storeB(0, breg0);
    asm volatile("s_wait_asynccnt 0x0" ::: "memory");
    __syncthreads();
    if (c + 1 < nch) { issueA(c + 1, 1); loadB(c + 1, breg1); }
    compute(0);
    if (++c >= nch) break;
    // odd step: chunk c in buffer 1
    storeB(1, breg1);
    asm volatile("s_wait_asynccnt 0x0" ::: "memory");
    __syncthreads();
    if (c + 1 < nch) { issueA(c + 1, 0); loadB(c + 1, breg0); }
    compute(1);
    if (++c >= nch) break;
  }

  // ---- Epilogue: BN + residual + ReLU, store f16 (predicated, post-WMMA) ----
#pragma unroll
  for (int nbk = 0; nbk < 2; ++nbk) {
    int n = n0 + wn * 32 + nbk * 16 + l16;
    if (n < Ntot) {
      int t = n / HWo;
      int r = n - t * HWo;
#pragma unroll
      for (int v = 0; v < 8; ++v) {
#pragma unroll
        for (int mbk = 0; mbk < 2; ++mbk) {
          int co = m0 + wm * 32 + mbk * 16 + half * 8 + v;
          float a = (mbk == 0) ? (nbk == 0 ? acc00[v] : acc01[v])
                               : (nbk == 0 ? acc10[v] : acc11[v]);
          float y = a * scale[co] + bias[co];
          size_t idx = ((size_t)t * Cout + co) * HWo + r;
          if (residual) y += (float)residual[idx];
          if (relu) y = fmaxf(y, 0.0f);
          out[idx] = (_Float16)y;
        }
      }
    }
  }
}

// ---------------------------------------------------------------------------
__global__ void fg_maxpool(const _Float16* __restrict__ in, _Float16* __restrict__ out,
                           int TC, int H, int W, int k, int s, int pad, int Ho, int Wo) {
  int i = blockIdx.x * blockDim.x + threadIdx.x;
  int total = TC * Ho * Wo;
  if (i >= total) return;
  int wo = i % Wo; int tmp = i / Wo; int ho = tmp % Ho; int tc = tmp / Ho;
  float m = -3.0e38f;
  for (int dy = 0; dy < k; ++dy) {
    int y = ho * s - pad + dy;
    if (y < 0 || y >= H) continue;
    for (int dx = 0; dx < k; ++dx) {
      int x = wo * s - pad + dx;
      if (x < 0 || x >= W) continue;
      m = fmaxf(m, (float)in[((size_t)tc * H + y) * W + x]);
    }
  }
  out[i] = (_Float16)m;
}

// ---------------------------------------------------------------------------
// linear_interp scans + box construction (T=64, P=3). Single-thread kernel.
// boxes per t (24 ints): hb[4] rb[4] lb[4] u_hl[4] u_hr[4] u_rl[4]; box=[x0,x1,y0,y1]
// ---------------------------------------------------------------------------
__global__ void fg_proposal(const float* __restrict__ ctr, const int* __restrict__ xlen,
                            int* __restrict__ boxes) {
  if (threadIdx.x != 0 || blockIdx.x != 0) return;
  const int T = 64;
  bool start[64], last[64];
  for (int t = 0; t < T; ++t) { start[t] = false; last[t] = false; }
  int cum = 0;
  for (int i = 0; i < 4; ++i) { start[cum] = true; cum += xlen[i]; last[cum - 1] = true; }

  int cx[3][64], cy[3][64];
  for (int p = 0; p < 3; ++p) {
    int prev[64], nxt[64];
    int c = -1;
    for (int t = 0; t < T; ++t) {
      if (start[t]) c = -1;
      if (ctr[(t * 3 + p) * 3 + 2] != 0.0f) c = t;
      prev[t] = c;
    }
    c = -1;
    for (int t = T - 1; t >= 0; --t) {
      if (last[t]) c = -1;
      if (ctr[(t * 3 + p) * 3 + 2] != 0.0f) c = t;
      nxt[t] = c;
    }
    for (int t = 0; t < T; ++t) {
      float ox, oy;
      if (ctr[(t * 3 + p) * 3 + 2] != 0.0f) {
        ox = ctr[(t * 3 + p) * 3 + 0];
        oy = ctr[(t * 3 + p) * 3 + 1];
      } else {
        int pv = prev[t] < 0 ? 0 : prev[t];
        int nx = nxt[t] < 0 ? 0 : nxt[t];
        float pfx = ctr[(pv * 3 + p) * 3 + 0], pfy = ctr[(pv * 3 + p) * 3 + 1];
        float pnx = ctr[(nx * 3 + p) * 3 + 0], pny = ctr[(nx * 3 + p) * 3 + 1];
        int dn = nxt[t] - prev[t]; if (dn < 1) dn = 1;
        float wgt = (float)(t - prev[t]) / (float)dn;
        ox = pfx + wgt * (pnx - pfx);
        oy = pfy + wgt * (pny - pfy);
        if (nxt[t] < 0)  { ox = pfx; oy = pfy; }
        if (prev[t] < 0) { ox = pnx; oy = pny; }
      }
      cx[p][t] = (int)floorf(ox * 0.25f);   // * (56/224)
      cy[p][t] = (int)floorf(oy * 0.25f);
    }
  }
  const int halfs[3] = {8, 12, 12};          // head, rhand, lhand
  for (int t = 0; t < T; ++t) {
    int b[3][4];
    for (int p = 0; p < 3; ++p) {
      b[p][0] = min(max(cx[p][t] - halfs[p], 0), 56);
      b[p][1] = min(max(cx[p][t] + halfs[p], 0), 56);
      b[p][2] = min(max(cy[p][t] - halfs[p], 0), 56);
      b[p][3] = min(max(cy[p][t] + halfs[p], 0), 56);
    }
    int* o = boxes + t * 24;
    for (int p = 0; p < 3; ++p)
      for (int k2 = 0; k2 < 4; ++k2) o[p * 4 + k2] = b[p][k2];
    const int ua[3] = {0, 0, 1}, ub[3] = {2, 1, 2};   // (h,l) (h,r) (r,l)
    for (int u = 0; u < 3; ++u) {
      int A = ua[u], B = ub[u];
      o[12 + u * 4 + 0] = min(b[A][0], b[B][0]);
      o[12 + u * 4 + 1] = max(b[A][1], b[B][1]);
      o[12 + u * 4 + 2] = min(b[A][2], b[B][2]);
      o[12 + u * 4 + 3] = max(b[A][3], b[B][3]);
    }
  }
}

// ---------------------------------------------------------------------------
// crop_align. mode 0=head(rb=1, cr=x0!=0) 1=rhand(rb=y0==0, cr=1) 2=lhand(rb=y0==0, cr=0)
// ---------------------------------------------------------------------------
__global__ void fg_crop(const _Float16* __restrict__ in, _Float16* __restrict__ out,
                        const int* __restrict__ boxes, int box_off, int G, int mode,
                        int T, int C, int H, int W) {
  int i = blockIdx.x * blockDim.x + threadIdx.x;
  int total = T * C * G * G;
  if (i >= total) return;
  int gx = i % G; int t2 = i / G; int gy = t2 % G; t2 /= G; int c = t2 % C; int t = t2 / C;
  const int* b = boxes + t * 24 + box_off;
  int x0 = b[0], x1 = b[1], y0 = b[2], y1 = b[3];
  bool rb = (mode == 0) ? true : (y0 == 0);
  bool cr = (mode == 0) ? (x0 != 0) : (mode == 1);
  int row = rb ? (y1 - G + gy) : (y0 + gy);
  int col = cr ? (x1 - G + gx) : (x0 + gx);
  float v = 0.0f;
  if (row >= y0 && row < y1 && col >= x0 && col < x1) {
    int rr = min(max(row, 0), H - 1);
    int cc = min(max(col, 0), W - 1);
    v = (float)in[(((size_t)t * C + c) * H + rr) * W + cc];
  }
  out[i] = (_Float16)v;
}

// ---------------------------------------------------------------------------
__global__ void fg_adaptive(const _Float16* __restrict__ in, _Float16* __restrict__ out,
                            const int* __restrict__ boxes, int box_off,
                            int T, int C, int H, int W) {
  const int G = 32, K = 4;
  int i = blockIdx.x * blockDim.x + threadIdx.x;
  int total = T * C * G * G;
  if (i >= total) return;
  int gx = i % G; int t2 = i / G; int gy = t2 % G; t2 /= G; int c = t2 % C; int t = t2 / C;
  const int* b = boxes + t * 24 + box_off;
  int x0 = b[0], x1 = b[1], y0 = b[2], y1 = b[3];
  int nr = y1 - y0, nc = x1 - x0;
  int sr = y0 + gy * nr / G, er = y0 + ((gy + 1) * nr + G - 1) / G;
  int sc = x0 + gx * nc / G, ec = x0 + ((gx + 1) * nc + G - 1) / G;
  float m = -3.0e38f;
  for (int ky = 0; ky < K; ++ky) {
    int row = sr + ky;
    if (row >= er) continue;
    int rr = min(max(row, 0), H - 1);
    for (int kx = 0; kx < K; ++kx) {
      int col = sc + kx;
      if (col >= ec) continue;
      int cc = min(max(col, 0), W - 1);
      m = fmaxf(m, (float)in[(((size_t)t * C + c) * H + rr) * W + cc]);
    }
  }
  out[i] = (_Float16)m;
}

// ---------------------------------------------------------------------------
__global__ void fg_mean_slot(const _Float16* __restrict__ in, float* __restrict__ out,
                             int T, int C, int HW, int slot) {
  int i = blockIdx.x * blockDim.x + threadIdx.x;
  if (i >= T * C) return;
  int t = i / C, c = i % C;
  const _Float16* p = in + (size_t)(t * C + c) * HW;
  float s = 0.0f;
  for (int k = 0; k < HW; ++k) s += (float)p[k];
  out[((size_t)t * 7 + slot) * C + c] = s / (float)HW;
}

// ===========================================================================
// Host orchestration
// ===========================================================================
extern "C" void kernel_launch(void* const* d_in, const int* in_sizes, int n_in,
                              void* d_out, int out_size, void* d_ws, size_t ws_size,
                              hipStream_t stream) {
  (void)in_sizes; (void)n_in; (void)out_size; (void)ws_size;
  const int T = 64;
  const float* x    = (const float*)d_in[0];
  const float* ctr  = (const float*)d_in[1];
  const int*   xlen = (const int*)d_in[2];
  float* out = (float*)d_out;

  size_t off = 0;
  auto alloc = [&](size_t bytes) -> char* {
    char* p = (char*)d_ws + off;
    off += (bytes + 255) & ~(size_t)255;
    return p;
  };

  // {w_idx, bn_g_idx, Cout, Cin, k} in flattened d_in order
  static const int UD[26][5] = {
    {3,4,64,3,7},
    {8,9,64,64,3},{13,14,64,64,3},{18,19,64,64,3},{23,24,64,64,3},
    {28,29,128,64,3},{33,34,128,128,3},{38,39,128,64,1},
    {43,44,128,128,3},{48,49,128,128,3},
    {53,54,256,128,3},{58,59,256,256,3},{63,64,256,128,1},
    {68,69,256,256,3},{73,74,256,256,3},
    {78,79,512,256,3},{83,84,512,512,3},{88,89,512,256,1},
    {93,94,512,512,3},{98,99,512,512,3},
    {103,104,128,64,5},{108,109,256,128,5},{113,114,512,256,5},
    {118,119,128,64,5},{123,124,256,128,5},{128,129,512,256,5}
  };
  struct Unit { int Cout, Cin, k; _Float16* wh; float* scale; float* bias; };
  Unit U[26];
  for (int u = 0; u < 26; ++u) {
    U[u].Cout = UD[u][2]; U[u].Cin = UD[u][3]; U[u].k = UD[u][4];
    size_t wn = (size_t)U[u].Cout * U[u].Cin * U[u].k * U[u].k;
    U[u].wh    = (_Float16*)alloc(wn * sizeof(_Float16));
    U[u].scale = (float*)alloc(U[u].Cout * sizeof(float));
    U[u].bias  = (float*)alloc(U[u].Cout * sizeof(float));
    fg_cvt_f32_f16<<<(unsigned)((wn + 255) / 256), 256, 0, stream>>>(
        (const float*)d_in[UD[u][0]], U[u].wh, (int)wn);
    fg_bn_prep<<<(U[u].Cout + 255) / 256, 256, 0, stream>>>(
        (const float*)d_in[UD[u][1]],     (const float*)d_in[UD[u][1] + 1],
        (const float*)d_in[UD[u][1] + 2], (const float*)d_in[UD[u][1] + 3],
        U[u].scale, U[u].bias, U[u].Cout);
  }

  size_t xh_n = (size_t)T * 3 * 224 * 224;
  _Float16* xh    = (_Float16*)alloc(xh_n * 2);
  _Float16* stem  = (_Float16*)alloc((size_t)T * 64 * 112 * 112 * 2);
  _Float16* P     = (_Float16*)alloc((size_t)T * 64 * 56 * 56 * 2);   // y
  _Float16* TA    = (_Float16*)alloc((size_t)T * 64 * 56 * 56 * 2);   // big temp
  _Float16* D0    = (_Float16*)alloc((size_t)T * 128 * 28 * 28 * 2);
  _Float16* D1    = (_Float16*)alloc((size_t)T * 128 * 28 * 28 * 2);
  _Float16* PB    = (_Float16*)alloc((size_t)T * 128 * 12 * 12 * 2);
  _Float16* headc = (_Float16*)alloc((size_t)T * 64 * 16 * 16 * 2);
  _Float16* rhc   = (_Float16*)alloc((size_t)T * 64 * 24 * 24 * 2);
  _Float16* lhc   = (_Float16*)alloc((size_t)T * 64 * 24 * 24 * 2);
  _Float16* ad0   = (_Float16*)alloc((size_t)T * 64 * 32 * 32 * 2);
  _Float16* ad1   = (_Float16*)alloc((size_t)T * 64 * 32 * 32 * 2);
  _Float16* ad2   = (_Float16*)alloc((size_t)T * 64 * 32 * 32 * 2);
  int* boxes      = (int*)alloc((size_t)T * 24 * sizeof(int));

  auto conv = [&](const _Float16* in, int u, _Float16* op, const _Float16* res,
                  int H, int W, int stride, int pad, int relu) {
    const Unit& un = U[u];
    int Ho = (H + 2 * pad - un.k) / stride + 1;
    int Wo = (W + 2 * pad - un.k) / stride + 1;
    dim3 grid((unsigned)((T * Ho * Wo + 127) / 128), (unsigned)(un.Cout / 64));
    fg_conv_wmma<<<grid, 256, 0, stream>>>(in, un.wh, un.scale, un.bias, res, op,
        T, un.Cin, H, W, un.Cout, Ho, Wo, un.k, un.k, stride, pad, relu);
  };
  auto pool = [&](const _Float16* in, _Float16* op, int C, int H, int W,
                  int k, int s, int pad) {
    int Ho = (H + 2 * pad - k) / s + 1;
    int Wo = (W + 2 * pad - k) / s + 1;
    int total = T * C * Ho * Wo;
    fg_maxpool<<<(total + 255) / 256, 256, 0, stream>>>(in, op, T * C, H, W, k, s, pad, Ho, Wo);
  };
  auto run_layer = [&](const _Float16* in, int H, int uc1, int uc2, int ud,
                       int ub1, int ub2, _Float16* tmp, _Float16* D) -> int {
    int Ho = H / 2;
    conv(in, ud, D, nullptr, H, H, 2, 0, 0);     // downsample (no relu)
    conv(in, uc1, tmp, nullptr, H, H, 2, 1, 1);  // conv1 + relu
    conv(tmp, uc2, D, D, Ho, Ho, 1, 1, 1);       // conv2 + res + relu (alias-safe)
    conv(D, ub1, tmp, nullptr, Ho, Ho, 1, 1, 1); // block1 conv1
    conv(tmp, ub2, D, D, Ho, Ho, 1, 1, 1);       // block1 conv2 + res
    return Ho;
  };
  auto tail = [&](const _Float16* in, int Hin, int slot) {
    int H2 = run_layer(in, Hin, 5, 6, 7, 8, 9, TA, D0);
    int H3 = run_layer(D0, H2, 10, 11, 12, 13, 14, TA, D1);
    int H4 = run_layer(D1, H3, 15, 16, 17, 18, 19, TA, D0);
    fg_mean_slot<<<(T * 512 + 255) / 256, 256, 0, stream>>>(D0, out, T, 512, H4 * H4, slot);
  };
  auto simple_cnn = [&](const _Float16* in, int G, int u0, int slot) {
    const _Float16* cur = in;
    int H = G;
    const int Cs[4] = {64, 128, 256, 512};
    for (int j = 0; j < 3; ++j) {
      conv(cur, u0 + j, TA, nullptr, H, H, 1, 2, 1);  // 5x5 pad2 + relu
      pool(TA, PB, Cs[j + 1], H, H, 2, 2, 0);
      H /= 2;
      cur = PB;
    }
    fg_mean_slot<<<(T * 512 + 255) / 256, 256, 0, stream>>>(PB, out, T, 512, H * H, slot);
  };

  // ---- pipeline ----
  fg_cvt_f32_f16<<<(unsigned)((xh_n + 255) / 256), 256, 0, stream>>>(x, xh, (int)xh_n);

  conv(xh, 0, stem, nullptr, 224, 224, 2, 3, 1);
  pool(stem, P, 64, 112, 112, 3, 2, 1);

  conv(P, 1, TA, nullptr, 56, 56, 1, 1, 1);
  conv(TA, 2, P, P, 56, 56, 1, 1, 1);
  conv(P, 3, TA, nullptr, 56, 56, 1, 1, 1);
  conv(TA, 4, P, P, 56, 56, 1, 1, 1);

  fg_proposal<<<1, 1, 0, stream>>>(ctr, xlen, boxes);
  fg_crop<<<(T * 64 * 16 * 16 + 255) / 256, 256, 0, stream>>>(P, headc, boxes, 0, 16, 0, T, 64, 56, 56);
  fg_crop<<<(T * 64 * 24 * 24 + 255) / 256, 256, 0, stream>>>(P, rhc,   boxes, 4, 24, 1, T, 64, 56, 56);
  fg_crop<<<(T * 64 * 24 * 24 + 255) / 256, 256, 0, stream>>>(P, lhc,   boxes, 8, 24, 2, T, 64, 56, 56);
  fg_adaptive<<<(T * 64 * 32 * 32 + 255) / 256, 256, 0, stream>>>(P, ad0, boxes, 12, T, 64, 56, 56);
  fg_adaptive<<<(T * 64 * 32 * 32 + 255) / 256, 256, 0, stream>>>(P, ad1, boxes, 16, T, 64, 56, 56);
  fg_adaptive<<<(T * 64 * 32 * 32 + 255) / 256, 256, 0, stream>>>(P, ad2, boxes, 20, T, 64, 56, 56);

  tail(P, 56, 0);
  simple_cnn(headc, 16, 20, 1);
  simple_cnn(lhc,   24, 23, 2);
  simple_cnn(rhc,   24, 23, 3);
  tail(ad0, 32, 4);
  tail(ad1, 32, 5);
  tail(ad2, 32, 6);
}
